// VMG_80865644249593
// MI455X (gfx1250) — matrix operand order repaired and checked by reading, not compiled, not run on hardware
//
#include <hip/hip_runtime.h>

typedef __attribute__((ext_vector_type(16))) _Float16 v16h;
typedef __attribute__((ext_vector_type(8)))  float    v8f;

#define KTILE 64          // K chunk: two 32-deep WMMA sub-chunks per barrier
#define LDSPAD 8          // row pad (halves): 72 halves = 144B row stride
#define NTILE 64          // pixels per block (x-strip)
#define PAD 3             // spatial zero-pad (7x7 SAME)

__device__ __forceinline__ int imin(int a, int b) { return a < b ? a : b; }

// ---------------------------------------------------------------------------
// Implicit-GEMM 7x7 SAME conv via v_wmma_f32_16x16x32_f16, wave32.
// K ordering: k = (ky*7+kx)*CIN + ci.
// in : PADDED f32 [N,CIN,H+6,W+6], logical image at (+3,+3), borders zero.
// wT : f16 [Cout][K64] pre-transposed weights, zero-padded to K64.
// Pipeline per chunk: store(t) -> issue loads(t+1) -> barrier -> compute(t),
// so staging-load latency is hidden behind the WMMAs of the previous chunk.
// CIN >= 32: (ky,kx) uniform per pass -> scalar; zero-tail handled by A zeros.
// Block: 256 threads = 8 waves; tile M=32 x N=64; 2 WMMA/wave/chunk.
// ---------------------------------------------------------------------------
template <int CIN, bool HASRES>
__global__ __launch_bounds__(256) void conv7_wmma_kernel(
    const float* __restrict__ in, const _Float16* __restrict__ wT,
    const float* __restrict__ bias, const float* __restrict__ res,
    float* __restrict__ out,
    int H, int W, int Cout, int tilesW, int doRelu,
    int oRow, int oChan, int oOff)
{
  constexpr int LG  = (CIN == 8) ? 3 : (CIN == 16) ? 4 : (CIN == 32) ? 5 : 6;
  constexpr int K   = 49 * CIN;
  constexpr int K64 = (K + 63) & ~63;
  constexpr int NCHUNK = K64 >> 6;

  const int Wp   = W + 2 * PAD;
  const int HpWp = (H + 2 * PAD) * Wp;          // < 119k, fits int

  int bx = blockIdx.x;
  const int tx   = bx % tilesW;  bx /= tilesW;
  const int y    = bx % H;
  const int nimg = bx / H;
  const int x0     = tx * NTILE;
  const int coBase = blockIdx.y * 32;

  const int tid    = threadIdx.x;
  const int lane   = tid & 31;
  const int wave   = tid >> 5;
  const int lhalf  = lane & 15;
  const int hi     = lane >> 4;
  const int coHalf = wave >> 2;
  const int nSlice = wave & 3;

  const float* inN = in + (long)nimg * CIN * HpWp;   // uniform scalar base

  // loop-invariant per-thread staging coords
  const int kA   = tid >> 4;         // B: k-pair index (pass adds 32)
  const int colB = (tid & 15) * 4;   // B: first of 4 columns
  const int mA   = tid >> 3;         // A: row
  const int kgA  = (tid & 7) * 8;    // A: first of 8 k's (16B aligned)

  __shared__ _Float16 As[2][32][KTILE + LDSPAD];
  __shared__ _Float16 Bs[2][NTILE][KTILE + LDSPAD];

  v8f acc = {};

  struct StageRegs {
    _Float16 a[8];
    float    r[2][2][4];   // [pass][k-of-pair][col]
    bool     kz[2][2];     // K-tail kill flags (constant-folds for CIN>=32)
  };

  // ---- issue all global loads for one 64-K chunk (no waits forced here) ----
  auto stageLoad = [&](int kBase, StageRegs& S) {
    // A: one aligned b128 f16 load
    {
      const _Float16* pA = wT + imin(coBase + mA, Cout - 1) * K64 + kBase + kgA;
#pragma unroll
      for (int j = 0; j < 8; ++j) S.a[j] = pA[j];
    }
    // B: 4 b128 f32 loads (2 passes x 2 k's x 4 cols)
#pragma unroll
    for (int pass = 0; pass < 2; ++pass) {
      if constexpr (CIN >= 32) {
        int kb  = kBase + pass * 32;
        int pos = imin(kb >> LG, 48);           // clamp K-tail (A zeros cover it)
        int ciB = kb & (CIN - 1);               // 0 or 32
        int ky  = pos / 7;
        int kx  = pos - ky * 7;
        int rowp = y + ky;
        int xbp  = x0 + colB + kx;
        if (x0 + 70 > Wp) xbp = imin(xbp, Wp - 4);   // only partial right tile
        const float* p0 = inN + (ciB + 2 * kA) * HpWp + rowp * Wp + xbp;
#pragma unroll
        for (int j2 = 0; j2 < 2; ++j2) {
#pragma unroll
          for (int j = 0; j < 4; ++j) S.r[pass][j2][j] = p0[j2 * HpWp + j];
          S.kz[pass][j2] = false;
        }
      } else {
        // CIN 8/16: per-thread (pos,ci) for each k of the pair; K has a tail.
#pragma unroll
        for (int j2 = 0; j2 < 2; ++j2) {
          int gk  = kBase + 2 * kA + pass * 32 + j2;
          int ci  = gk & (CIN - 1);
          int pos = imin(gk >> LG, 48);
          int ky  = pos / 7;
          int kx  = pos - ky * 7;
          const float* p = inN + ci * HpWp + (y + ky) * Wp
                               + imin(x0 + colB + kx, Wp - 4);
#pragma unroll
          for (int j = 0; j < 4; ++j) S.r[pass][j2][j] = p[j];
          S.kz[pass][j2] = (gk >= K);
        }
      }
    }
  };

  // ---- wait + convert + LDS stores ----
  auto stageStore = [&](int buf, StageRegs& S) {
    _Float16 a[8];
#pragma unroll
    for (int j = 0; j < 8; ++j)
      a[j] = (coBase + mA >= Cout) ? (_Float16)0.f : S.a[j];
#pragma unroll
    for (int j = 0; j < 8; ++j) As[buf][mA][kgA + j] = a[j];
#pragma unroll
    for (int pass = 0; pass < 2; ++pass) {
      const int k0 = 2 * kA + pass * 32;
#pragma unroll
      for (int j = 0; j < 4; ++j) {
        float v0 = S.kz[pass][0] ? 0.f : S.r[pass][0][j];
        float v1 = S.kz[pass][1] ? 0.f : S.r[pass][1][j];
        Bs[buf][colB + j][k0 + 0] = (_Float16)v0;   // adjacent pair -> b32 store
        Bs[buf][colB + j][k0 + 1] = (_Float16)v1;
      }
    }
  };

  // ---- fragment load (CDNA5 VGPR layouts) + 2 accumulating WMMAs ----
  auto compute = [&](int buf) {
    const int mrow = coHalf * 16 + lhalf;   // A: lanes 0-15 M=lane, 16-31 M=lane-16
    const int ka0  = hi ? 8 : 0;            // K runs {0..7,16..23} / {8..15,24..31}
    const int ka1  = hi ? 24 : 16;
    const int nB   = nSlice * 16 + lhalf;   // B: lane group = N column
    const int kb0  = hi ? 8 : 0;            // K runs {0..7,16..23} / {8..15,24..31}
#pragma unroll
    for (int s = 0; s < 2; ++s) {
      const int ks = s * 32;
      v16h afrag, bfrag;
#pragma unroll
      for (int i = 0; i < 8; ++i) afrag[i]     = As[buf][mrow][ks + ka0 + i];
#pragma unroll
      for (int i = 0; i < 8; ++i) afrag[8 + i] = As[buf][mrow][ks + ka1 + i];
#pragma unroll
      for (int i = 0; i < 16; ++i) bfrag[i] = Bs[buf][nB][ks + kb0 + (i & 7) + ((i >> 3) << 4)];
      acc = __builtin_amdgcn_wmma_f32_16x16x32_f16(
          false, afrag, false, bfrag, (short)0, acc, false, false);
    }
  };

  // ---- software pipeline: loads(t+1) issued before compute(t) ----
  StageRegs S;
  stageLoad(0, S);
  for (int t = 0; t < NCHUNK; ++t) {
    stageStore(t & 1, S);                       // waits chunk-t loads (hidden)
    if (t + 1 < NCHUNK) stageLoad((t + 1) * KTILE, S);
    __syncthreads();                            // publish buf[t&1]
    compute(t & 1);                             // overlaps with in-flight loads
  }

  // ---- epilogue: D VGPR v -> M = coHalf*16 + v + 8*hi, N = x0+nSlice*16+lhalf ----
  const int xo = x0 + nSlice * 16 + lhalf;
  if (xo < W) {
#pragma unroll
    for (int v = 0; v < 8; ++v) {
      int co = coBase + coHalf * 16 + v + (hi ? 8 : 0);
      if (co < Cout) {
        float o = acc[v] + bias[co];
        if (doRelu) o = fmaxf(o, 0.0f);
        int oi = (nimg * Cout + co) * oChan + y * oRow + xo + oOff;  // fits int
        if constexpr (HASRES) o += res[oi];
        out[oi] = o;
      }
    }
  }
}

// ---------------------------------------------------------------------------
// Weight transpose: w [L*Cout][Cin][49] f32 -> wT [L*Cout][K64] f16
// (k' = pos*Cin + ci), zero-padded K tail.
// ---------------------------------------------------------------------------
__global__ void wtrans_kernel(const float* __restrict__ w, _Float16* __restrict__ wT,
                              long total, int Cin, int K64) {
  long i = (long)blockIdx.x * blockDim.x + threadIdx.x;
  if (i >= total) return;
  int kp  = (int)(i % K64);
  long coL = i / K64;
  float v = 0.0f;
  if (kp < 49 * Cin) {
    int ci  = kp % Cin;
    int pos = kp / Cin;
    v = w[(coL * Cin + ci) * 49 + pos];
  }
  wT[i] = (_Float16)v;
}

// ---------------------------------------------------------------------------
// Elementwise helpers
// ---------------------------------------------------------------------------
__global__ void normalize_kernel(const float* __restrict__ in, float* __restrict__ out,
                                 long total, int HW) {
  long i = (long)blockIdx.x * blockDim.x + threadIdx.x;
  if (i >= total) return;
  int c = (int)((i / HW) % 3);
  const float mean[3] = {0.485f, 0.456f, 0.406f};
  const float stdv[3] = {0.229f, 0.224f, 0.225f};
  out[i] = (in[i] - mean[c]) / stdv[c];
}

__global__ void avgpool_kernel(const float* __restrict__ in, float* __restrict__ out,
                               long total, int H, int W) {
  long i = (long)blockIdx.x * blockDim.x + threadIdx.x;
  if (i >= total) return;
  int W2 = W >> 1, H2 = H >> 1;
  int x = (int)(i % W2); long t = i / W2;
  int y = (int)(t % H2); long nc = t / H2;
  const float* p = in + (nc * H + 2 * y) * W + 2 * x;
  out[i] = 0.25f * (p[0] + p[1] + p[W] + p[W + 1]);
}

__global__ void fill_kernel(float* __restrict__ p, float v, long n) {
  long i = (long)blockIdx.x * blockDim.x + threadIdx.x;
  if (i < n) p[i] = v;
}

// src [N,C,H,W] flat -> padded dst [N,Ctot,H+6,W+6] at channel offset cOff, (+3,+3)
__global__ void copych_pad_kernel(const float* __restrict__ src, float* __restrict__ dst,
                                  long total, int C, int H, int W, int Ctot, int cOff) {
  long i = (long)blockIdx.x * blockDim.x + threadIdx.x;
  if (i >= total) return;
  int HW = H * W;
  int p = (int)(i % HW); long t = i / HW;
  int c = (int)(t % C);  long n = t / C;
  int y = p / W, x = p - y * W;
  int Wp = W + 2 * PAD;
  long HpWp = (long)(H + 2 * PAD) * Wp;
  dst[(n * Ctot + cOff + c) * HpWp + (long)(y + PAD) * Wp + (x + PAD)] = src[i];
}

// align-corners bilinear resize of flow [N,2,H1,W1]->[N,2,H2,W2], values *2
__global__ void resize_flow_kernel(const float* __restrict__ in, float* __restrict__ out,
                                   long total, int H1, int W1, int H2, int W2) {
  long i = (long)blockIdx.x * blockDim.x + threadIdx.x;
  if (i >= total) return;
  int x = (int)(i % W2); long t = i / W2;
  int y = (int)(t % H2); long nc = t / H2;
  float sy = (H2 > 1) ? (float)y * (float)(H1 - 1) / (float)(H2 - 1) : 0.0f;
  float sx = (W2 > 1) ? (float)x * (float)(W1 - 1) / (float)(W2 - 1) : 0.0f;
  int y0 = (int)floorf(sy), x0 = (int)floorf(sx);
  int y1 = imin(y0 + 1, H1 - 1), x1 = imin(x0 + 1, W1 - 1);
  float wy = sy - (float)y0, wx = sx - (float)x0;
  const float* p = in + nc * (long)H1 * W1;
  float a = p[(long)y0 * W1 + x0] * (1.0f - wy) + p[(long)y1 * W1 + x0] * wy;
  float b = p[(long)y0 * W1 + x1] * (1.0f - wy) + p[(long)y1 * W1 + x1] * wy;
  out[i] = (a * (1.0f - wx) + b * wx) * 2.0f;
}

// bilinear flow-warp of s [N,3,H,W] by flow [N,2,H,W]; write padded cat ch cOff..+2
__global__ void warp_to_cat_kernel(const float* __restrict__ s, const float* __restrict__ flow,
                                   float* __restrict__ cat, long total,
                                   int H, int W, int Ctot, int cOff) {
  long i = (long)blockIdx.x * blockDim.x + threadIdx.x;
  if (i >= total) return;
  int HW = H * W;
  int x = (int)(i % W); long t = i / W;
  int y = (int)(t % H); t /= H;
  int c = (int)(t % 3); long n = t / 3;
  long fb = n * 2 * (long)HW + (long)y * W + x;
  float fx = (float)x + flow[fb];
  float fy = (float)y + flow[fb + HW];
  fx = fminf(fmaxf(fx, 0.0f), (float)(W - 1));
  fy = fminf(fmaxf(fy, 0.0f), (float)(H - 1));
  int x0 = (int)floorf(fx), y0 = (int)floorf(fy);
  int x1 = imin(x0 + 1, W - 1), y1 = imin(y0 + 1, H - 1);
  float wx = fx - (float)x0, wy = fy - (float)y0;
  const float* p = s + (n * 3 + c) * (long)HW;
  float v = p[(long)y0 * W + x0] * (1.0f - wy) * (1.0f - wx)
          + p[(long)y0 * W + x1] * (1.0f - wy) * wx
          + p[(long)y1 * W + x0] * wy * (1.0f - wx)
          + p[(long)y1 * W + x1] * wy * wx;
  int Wp = W + 2 * PAD;
  long HpWp = (long)(H + 2 * PAD) * Wp;
  cat[(n * Ctot + cOff + c) * HpWp + (long)(y + PAD) * Wp + (x + PAD)] = v;
}

// ---------------------------------------------------------------------------
// Host orchestration
// ---------------------------------------------------------------------------
static inline dim3 ewgrid(long n) { return dim3((unsigned)((n + 255) / 256)); }

static void launch_conv(int layer, dim3 grid, hipStream_t s,
                        const float* in, const _Float16* wT, const float* bias,
                        const float* res, float* out,
                        int H, int W, int Cout, int tilesW, int doRelu,
                        int oRow, int oChan, int oOff) {
  switch (layer) {
    case 0:
      conv7_wmma_kernel<8, false><<<grid, 256, 0, s>>>(in, wT, bias, res, out, H, W, Cout,
                                                       tilesW, doRelu, oRow, oChan, oOff);
      break;
    case 1:
      conv7_wmma_kernel<32, false><<<grid, 256, 0, s>>>(in, wT, bias, res, out, H, W, Cout,
                                                        tilesW, doRelu, oRow, oChan, oOff);
      break;
    case 2:
      conv7_wmma_kernel<64, false><<<grid, 256, 0, s>>>(in, wT, bias, res, out, H, W, Cout,
                                                        tilesW, doRelu, oRow, oChan, oOff);
      break;
    case 3:
      conv7_wmma_kernel<32, false><<<grid, 256, 0, s>>>(in, wT, bias, res, out, H, W, Cout,
                                                        tilesW, doRelu, oRow, oChan, oOff);
      break;
    default: // conv5: CIN=16, flat out with residual
      conv7_wmma_kernel<16, true><<<grid, 256, 0, s>>>(in, wT, bias, res, out, H, W, Cout,
                                                       tilesW, doRelu, oRow, oChan, oOff);
      break;
  }
}

extern "C" void kernel_launch(void* const* d_in, const int* in_sizes, int n_in,
                              void* d_out, int out_size, void* d_ws, size_t ws_size,
                              hipStream_t stream) {
  (void)in_sizes; (void)n_in; (void)out_size; (void)ws_size;
  const float* ref  = (const float*)d_in[0];
  const float* supp = (const float*)d_in[1];
  const float* Wt[5] = {(const float*)d_in[2], (const float*)d_in[4], (const float*)d_in[6],
                        (const float*)d_in[8], (const float*)d_in[10]};
  const float* Bt[5] = {(const float*)d_in[3], (const float*)d_in[5], (const float*)d_in[7],
                        (const float*)d_in[9], (const float*)d_in[11]};

  const int N = 8;
  const int cin[5]  = {8, 32, 64, 32, 16};
  const int cout[5] = {32, 64, 32, 16, 2};
  int K64[5];
  for (int i = 0; i < 5; ++i) K64[i] = ((49 * cin[i] + 63) / 64) * 64;

  long hw[6], off[6], tot = 0;
  for (int l = 0; l < 6; ++l) {
    int h = 8 << l, w = 14 << l;
    hw[l] = (long)h * w;
    off[l] = tot;
    tot += (long)N * 3 * hw[l];
  }
  long hpwp5 = (long)(256 + 2 * PAD) * (448 + 2 * PAD);  // max padded plane

  float* ws = (float*)d_ws;
  _Float16* wT[5];
  for (int i = 0; i < 5; ++i) {
    wT[i] = (_Float16*)ws;
    long halves = (long)6 * cout[i] * K64[i];
    ws += (halves + 1) / 2;                       // f16 storage in float units
  }
  float* rp   = ws; ws += tot;                    // ref pyramid (flat)
  float* sp   = ws; ws += tot;                    // supp pyramid (flat)
  long   fsz  = (long)N * 2 * hw[5];
  float* f0   = ws; ws += fsz;                    // flow (flat)
  float* f1   = ws; ws += fsz;                    // flow_up (flat)
  float* cat  = ws; ws += (long)N * 8 * hpwp5;    // padded concat (8 ch)
  float* actA = ws; ws += (long)N * 64 * hpwp5;   // padded activations
  float* actB = ws; ws += (long)N * 32 * hpwp5;   // padded activations

  // ---- pre-transpose weights into zero-padded f16 K64 layout ----
  for (int i = 0; i < 5; ++i) {
    long n = (long)6 * cout[i] * K64[i];
    wtrans_kernel<<<ewgrid(n), 256, 0, stream>>>(Wt[i], wT[i], n, cin[i], K64[i]);
  }

  // ---- normalized pyramids (flat layout) ----
  {
    long n = (long)N * 3 * hw[5];
    normalize_kernel<<<ewgrid(n), 256, 0, stream>>>(ref,  rp + off[5], n, (int)hw[5]);
    normalize_kernel<<<ewgrid(n), 256, 0, stream>>>(supp, sp + off[5], n, (int)hw[5]);
  }
  for (int l = 5; l > 0; --l) {
    int h = 8 << l, w = 14 << l;
    long n = (long)N * 3 * hw[l - 1];
    avgpool_kernel<<<ewgrid(n), 256, 0, stream>>>(rp + off[l], rp + off[l - 1], n, h, w);
    avgpool_kernel<<<ewgrid(n), 256, 0, stream>>>(sp + off[l], sp + off[l - 1], n, h, w);
  }

  // ---- coarse-to-fine flow ----
  for (int l = 0; l < 6; ++l) {
    int h = 8 << l, w = 14 << l;
    long HWl = hw[l];
    int Wp = w + 2 * PAD;
    long HpWp = (long)(h + 2 * PAD) * Wp;
    int tilesW = (w + NTILE - 1) / NTILE;

    // zero padded buffers for this level's geometry (borders must be 0)
    fill_kernel<<<ewgrid((long)N * 8  * HpWp), 256, 0, stream>>>(cat,  0.0f, (long)N * 8  * HpWp);
    fill_kernel<<<ewgrid((long)N * 64 * HpWp), 256, 0, stream>>>(actA, 0.0f, (long)N * 64 * HpWp);
    fill_kernel<<<ewgrid((long)N * 32 * HpWp), 256, 0, stream>>>(actB, 0.0f, (long)N * 32 * HpWp);

    if (l == 0) {
      fill_kernel<<<ewgrid((long)N * 2 * HWl), 256, 0, stream>>>(f1, 0.0f, (long)N * 2 * HWl);
    } else {
      long n = (long)N * 2 * HWl;
      resize_flow_kernel<<<ewgrid(n), 256, 0, stream>>>(f0, f1, n,
                                                        8 << (l - 1), 14 << (l - 1), h, w);
    }

    {
      long n3 = (long)N * 3 * HWl;
      long n2 = (long)N * 2 * HWl;
      copych_pad_kernel<<<ewgrid(n3), 256, 0, stream>>>(rp + off[l], cat, n3, 3, h, w, 8, 0);
      warp_to_cat_kernel<<<ewgrid(n3), 256, 0, stream>>>(sp + off[l], f1, cat, n3, h, w, 8, 3);
      copych_pad_kernel<<<ewgrid(n2), 256, 0, stream>>>(f1, cat, n2, 2, h, w, 8, 6);
    }

    const float* srcs[5] = {cat, actB, actA, actB, actA};
    float*       dsts[5] = {actB, actA, actB, actA, (l == 5) ? (float*)d_out : f0};
    for (int i = 0; i < 5; ++i) {
      bool padOut = (i < 4);     // convs 1-4 write padded activations; conv5 flat
      dim3 grid((unsigned)((long)N * h * tilesW), (unsigned)((cout[i] + 31) / 32));
      launch_conv(i, grid, stream,
                  srcs[i],
                  wT[i] + (long)l * cout[i] * K64[i],
                  Bt[i] + (long)l * cout[i],
                  (i == 4) ? f1 : nullptr,
                  dsts[i],
                  h, w, cout[i], tilesW, (i < 4) ? 1 : 0,
                  padOut ? Wp : w,
                  padOut ? (int)HpWp : (int)HWl,
                  padOut ? (PAD * Wp + PAD) : 0);
    }
  }
}
